// RallyNet_42494406427514
// MI455X (gfx1250) — matrix operand order repaired and checked by reading, not compiled, not run on hardware
//
#include <hip/hip_runtime.h>

// ---------------------------------------------------------------------------
// RallyNet decoder forward for MI455X (gfx1250, wave32, WMMA).
//
// Roofline: ~45 GF of GEMM vs ~125 MB HBM traffic (AI ~360 FLOP/B) -> matrix-
// core work.  bf16 WMMA (16x16x32, K=32/instr) gives 8x the per-instruction
// throughput of the f32 WMMA path with f32 accumulation; inputs are small-
// scale Gaussians so bf16 A/B is the precision the chip rewards.
//
// GEMM kernel: BM=256 x BN=64 block, 8 waves, each wave owns 32x64
// (2 A-frags x 4 B-frags = 8 v_wmma per K-step, B-frags reused).  Double-
// buffered LDS + register-staged software pipeline: global loads for tile
// k+1 issue before the 8 WMMAs of tile k, the loadcnt wait lands after them.
// f32->bf16 staging uses v_perm_b32 packing (3 VALU per pair).
// ---------------------------------------------------------------------------

typedef __attribute__((ext_vector_type(16))) __bf16 v16bf;
typedef __attribute__((ext_vector_type(8)))  float  v8f;

#define BDIM  4096
#define TDIM  35
#define DSE   227
#define CDIM  64
#define HDIM  1024
#define SE_NC 163
#define OUTW  415

#define BM 256
#define BN 64
#define BK 32
#define LSA 36            // padded LDS row stride (bf16 elems): 72B, 8B-aligned rows
#define LSB 36

union FragBF { v16bf v; unsigned u[8]; };

// f32 -> bf16 with round-half-up; single-value and v_perm-packed pair forms.
__device__ __forceinline__ unsigned short f2bf(float f) {
  unsigned u = __builtin_bit_cast(unsigned, f);
  return (unsigned short)((u + 0x8000u) >> 16);
}
__device__ __forceinline__ unsigned pack2bf(float lo, float hi) {
  unsigned ul = __builtin_bit_cast(unsigned, lo) + 0x8000u;
  unsigned uh = __builtin_bit_cast(unsigned, hi) + 0x8000u;
  // bytes: {uh[3],uh[2],ul[3],ul[2]}  -> one v_perm_b32
  return __builtin_amdgcn_perm(uh, ul, 0x07060302u);
}
__device__ __forceinline__ float softplus_f(float x) {
  return (x > 20.f) ? x : __logf(1.f + __expf(x));
}
__device__ __forceinline__ v8f wmma_bf16(const FragBF& a, const FragBF& b, v8f c) {
  return __builtin_amdgcn_wmma_f32_16x16x32_bf16(false, a.v, false, b.v,
                                                 (short)0, c, false, false);
}

// C[M,N] = act(A[M,K] @ W[K,N] + bias)
// MODE 0: A is a plain row-major f32 matrix (lda == K).
// MODE 1: A is the virtual sre = concat(state_embedding[b,t,:163], prior_z[b,:64])
//         flattened to K = T*DSE = 7945; per-thread (t,j) advanced incrementally.
template <int MODE>
__global__ __launch_bounds__(256)
void wmma_gemm_kernel(const float* __restrict__ A,
                      const float* __restrict__ W,
                      const float* __restrict__ bias,
                      float* __restrict__ Cout,
                      int M, int N, int K, int act,
                      const float* __restrict__ se,
                      const float* __restrict__ pz)
{
  __shared__ unsigned short As[2][BM * LSA];   // 36 KB
  __shared__ unsigned short Bs[2][BN * LSB];   //  9 KB

  const int tid    = threadIdx.x;
  const int wave   = tid >> 5;
  const int lane   = tid & 31;
  const int lane15 = lane & 15;
  const int laneHi = lane >> 4;
  const int m0     = blockIdx.x * BM;
  const int n0     = blockIdx.y * BN;

  v8f acc[2][4] = {};

  // per-thread staging registers (tile k+1 lives here while tile k computes)
  float a_reg[32];
  float b_reg[8];

  // MODE 1 incremental gather state: thread's K-column is fixed (k = tid&31),
  // so exactly one (t, j) pair per thread per tile; advance by 32 with wrap.
  int tt = 0, jj = tid & 31;

  const int ksteps = (K + BK - 1) / BK;

  // ---- staging helpers ----
  auto loadA = [&](int kb) {
    if (MODE == 0) {
      const int k2 = (tid & 15) << 1;           // thread owns K-pair
      const int gk = kb + k2;
      #pragma unroll
      for (int i = 0; i < 16; ++i) {
        int r = (i << 4) + (tid >> 4);
        const float* row = A + (size_t)(m0 + r) * K;
        a_reg[2 * i]     = (gk     < K) ? row[gk]     : 0.f;
        a_reg[2 * i + 1] = (gk + 1 < K) ? row[gk + 1] : 0.f;
      }
    } else {
      const int gk = kb + (tid & 31);
      const int t = tt, j = jj;
      #pragma unroll
      for (int i = 0; i < 32; ++i) {
        int r = (i << 3) + (tid >> 5);
        float v = 0.f;
        if (gk < K) {
          v = (j < SE_NC)
                ? se[((size_t)(m0 + r) * TDIM + t) * SE_NC + j]
                : pz[(size_t)(m0 + r) * CDIM + (j - SE_NC)];
        }
        a_reg[i] = v;
      }
      jj += 32; if (jj >= DSE) { jj -= DSE; ++tt; }
    }
  };
  auto loadB = [&](int kb) {
    const int n  = tid & 63;
    const int gn = n0 + n;
    #pragma unroll
    for (int i = 0; i < 4; ++i) {
      int k2 = ((i << 2) + (tid >> 6)) << 1;    // thread owns K-pair, fixed n
      int gk = kb + k2;
      bool on = (gn < N);
      b_reg[2 * i]     = (on && gk     < K) ? W[(size_t)gk * N + gn]       : 0.f;
      b_reg[2 * i + 1] = (on && gk + 1 < K) ? W[(size_t)(gk + 1) * N + gn] : 0.f;
    }
  };
  auto storeA = [&](int buf) {
    if (MODE == 0) {
      const int k2 = (tid & 15) << 1;
      #pragma unroll
      for (int i = 0; i < 16; ++i) {
        int r = (i << 4) + (tid >> 4);
        *(unsigned*)&As[buf][r * LSA + k2] = pack2bf(a_reg[2 * i], a_reg[2 * i + 1]);
      }
    } else {
      const int k = tid & 31;
      #pragma unroll
      for (int i = 0; i < 32; ++i) {
        int r = (i << 3) + (tid >> 5);
        As[buf][r * LSA + k] = f2bf(a_reg[i]);
      }
    }
  };
  auto storeB = [&](int buf) {
    const int n = tid & 63;
    #pragma unroll
    for (int i = 0; i < 4; ++i) {
      int k2 = ((i << 2) + (tid >> 6)) << 1;
      *(unsigned*)&Bs[buf][n * LSB + k2] = pack2bf(b_reg[2 * i], b_reg[2 * i + 1]);
    }
  };

  // ---- pipeline prologue ----
  loadA(0); loadB(0);
  storeA(0); storeB(0);

  for (int ks = 0; ks < ksteps; ++ks) {
    const int cur = ks & 1, nxt = cur ^ 1;
    __syncthreads();                                   // tile `cur` visible

    if (ks + 1 < ksteps) { loadA((ks + 1) * BK); loadB((ks + 1) * BK); }

    // ---- compute tile `cur`: fragments per ISA 7.12.2 wave32 layouts ----
    const unsigned short* Ac = As[cur];
    const unsigned short* Bc = Bs[cur];
    const int arow = (wave << 5) + lane15;             // wave owns 32 rows
    const int akb  = laneHi << 3;                      // 0 or 8
    FragBF af0, af1;
    #pragma unroll
    for (int j = 0; j < 8; ++j) {
      int k0 = ((j & 4) << 2) + akb + ((j & 3) << 1);  // {0..7,16..23}(+akb)
      af0.u[j] = *(const unsigned*)&Ac[arow * LSA + k0];
      af1.u[j] = *(const unsigned*)&Ac[(arow + 16) * LSA + k0];
    }
    const int bkb = laneHi << 3;                       // 0 or 8
    #pragma unroll
    for (int nt = 0; nt < 4; ++nt) {
      FragBF bfr;
      const int bcol = (nt << 4) + lane15;
      #pragma unroll
      for (int j = 0; j < 8; ++j)
        bfr.u[j] = *(const unsigned*)&Bc[bcol * LSB + ((j & 4) << 2) + bkb + ((j & 3) << 1)];
      acc[0][nt] = wmma_bf16(af0, bfr, acc[0][nt]);
      acc[1][nt] = wmma_bf16(af1, bfr, acc[1][nt]);    // B-frag reused
    }

    if (ks + 1 < ksteps) { storeA(nxt); storeB(nxt); } // loadcnt wait lands here
  }

  // ---- epilogue: D layout -> VGPR r = (M = r + 8*(lane/16), N = lane%16) ----
  const int mbase = m0 + (wave << 5) + (laneHi << 3);
  const int nbase = n0 + lane15;
  #pragma unroll
  for (int half = 0; half < 2; ++half) {
    #pragma unroll
    for (int nt = 0; nt < 4; ++nt) {
      int n = nbase + (nt << 4);
      if (n < N) {
        float bb = bias ? bias[n] : 0.f;
        #pragma unroll
        for (int r = 0; r < 8; ++r) {
          int m = mbase + (half << 4) + r;
          if (m < M) {
            float v = acc[half][nt][r] + bb;
            if (act == 1) v = softplus_f(v);
            Cout[(size_t)m * N + n] = v;
          }
        }
      }
    }
  }
}

// GMM / shot heads + log_softmax + ctx passthrough. One block per batch row.
__global__ __launch_bounds__(128)
void heads_kernel(const float* __restrict__ fl_g, const float* __restrict__ ctx_g,
                  const float* __restrict__ W_land, const float* __restrict__ W_shot,
                  const float* __restrict__ W_move,
                  const float* __restrict__ cW_land, const float* __restrict__ cW_shot,
                  const float* __restrict__ cW_move,
                  float* __restrict__ out)
{
  __shared__ float fl[DSE];
  __shared__ float cx[CDIM];
  __shared__ float slog[12], cslog[12];
  __shared__ float snorm, csnorm;

  const int b = blockIdx.x, tid = threadIdx.x;
  for (int i = tid; i < DSE; i += 128) fl[i] = fl_g[(size_t)b * DSE + i];
  if (tid < CDIM) cx[tid] = ctx_g[(size_t)b * CDIM + tid];
  __syncthreads();

  float* o = out + (size_t)b * OUTW;
  if (tid < 25) {                                    // land -> [0,25)
    float a = 0.f;
    for (int k = 0; k < DSE; ++k) a = fmaf(fl[k], W_land[k * 25 + tid], a);
    o[tid] = a;
  } else if (tid < 37) {                             // shot logits
    int j = tid - 25; float a = 0.f;
    for (int k = 0; k < DSE; ++k) a = fmaf(fl[k], W_shot[k * 12 + j], a);
    slog[j] = a;
  } else if (tid < 62) {                             // move -> [37,62)
    int j = tid - 37; float a = 0.f;
    for (int k = 0; k < DSE; ++k) a = fmaf(fl[k], W_move[k * 25 + j], a);
    o[37 + j] = a;
  } else if (tid >= 64 && tid < 89) {                // c_land -> [289,314)
    int j = tid - 64; float a = 0.f;
    for (int k = 0; k < CDIM; ++k) a = fmaf(cx[k], cW_land[k * 25 + j], a);
    o[289 + j] = a;
  } else if (tid >= 89 && tid < 101) {               // c_shot logits
    int j = tid - 89; float a = 0.f;
    for (int k = 0; k < CDIM; ++k) a = fmaf(cx[k], cW_shot[k * 12 + j], a);
    cslog[j] = a;
  } else if (tid >= 101 && tid < 126) {              // c_move -> [326,351)
    int j = tid - 101; float a = 0.f;
    for (int k = 0; k < CDIM; ++k) a = fmaf(cx[k], cW_move[k * 25 + j], a);
    o[326 + j] = a;
  }
  __syncthreads();
  if (tid == 0) {
    float m = slog[0];
    for (int j = 1; j < 12; ++j) m = fmaxf(m, slog[j]);
    float s = 0.f;
    for (int j = 0; j < 12; ++j) s += __expf(slog[j] - m);
    snorm = m + __logf(s);
  } else if (tid == 1) {
    float m = cslog[0];
    for (int j = 1; j < 12; ++j) m = fmaxf(m, cslog[j]);
    float s = 0.f;
    for (int j = 0; j < 12; ++j) s += __expf(cslog[j] - m);
    csnorm = m + __logf(s);
  }
  __syncthreads();
  if (tid < 12)       o[25  + tid]        = slog[tid] - snorm;        // shot
  else if (tid < 24)  o[314 + (tid - 12)] = cslog[tid - 12] - csnorm; // c_shot
  if (tid < CDIM)     o[351 + tid]        = cx[tid];                  // ctx
}

// g-nets: one wave32 per (b,d): g_out = sigmoid( sum_h sp(x*w1+b1)*w2 + b2 ).
__global__ __launch_bounds__(256)
void gnet_kernel(const float* __restrict__ fl,
                 const float* __restrict__ gW1, const float* __restrict__ gb1,
                 const float* __restrict__ gW2, const float* __restrict__ gb2,
                 float* __restrict__ out, int total)
{
  const int task = blockIdx.x * 8 + (threadIdx.x >> 5);
  const int lane = threadIdx.x & 31;
  if (task >= total) return;
  const int d = task % DSE;
  const int b = task / DSE;
  const float x = fl[(size_t)b * DSE + d];
  const float* w1 = gW1 + (size_t)d * DSE;
  const float* b1 = gb1 + (size_t)d * DSE;
  const float* w2 = gW2 + (size_t)d * DSE;
  float acc = 0.f;
  for (int h = lane; h < DSE; h += 32)
    acc = fmaf(softplus_f(fmaf(x, w1[h], b1[h])), w2[h], acc);
  #pragma unroll
  for (int off = 16; off; off >>= 1) acc += __shfl_xor(acc, off, 32);
  if (lane == 0) {
    float z = acc + gb2[d];
    out[(size_t)b * OUTW + 62 + d] = 1.f / (1.f + __expf(-z));  // g_out [62,289)
  }
}

extern "C" void kernel_launch(void* const* d_in, const int* in_sizes, int n_in,
                              void* d_out, int out_size, void* d_ws, size_t ws_size,
                              hipStream_t stream) {
  (void)in_sizes; (void)n_in; (void)out_size; (void)ws_size;
  const float* se     = (const float*)d_in[0];
  const float* pz     = (const float*)d_in[1];
  const float* W_ref  = (const float*)d_in[2];
  const float* fW1    = (const float*)d_in[3];
  const float* fb1    = (const float*)d_in[4];
  const float* fW2    = (const float*)d_in[5];
  const float* fb2    = (const float*)d_in[6];
  const float* fW3    = (const float*)d_in[7];
  const float* fb3    = (const float*)d_in[8];
  const float* gW1    = (const float*)d_in[9];
  const float* gb1    = (const float*)d_in[10];
  const float* gW2    = (const float*)d_in[11];
  const float* gb2    = (const float*)d_in[12];
  const float* W_land = (const float*)d_in[13];
  const float* W_shot = (const float*)d_in[14];
  const float* W_move = (const float*)d_in[15];
  const float* cW1    = (const float*)d_in[16];
  const float* cb1    = (const float*)d_in[17];
  const float* cW2    = (const float*)d_in[18];
  const float* cb2    = (const float*)d_in[19];
  const float* cW3    = (const float*)d_in[20];
  const float* cb3    = (const float*)d_in[21];
  const float* cW_land= (const float*)d_in[22];
  const float* cW_shot= (const float*)d_in[23];
  const float* cW_move= (const float*)d_in[24];
  float* out = (float*)d_out;

  // workspace layout (f32): 2*B*H ping-pong + ctx + f_input + f_latent (~40 MB)
  float* ws   = (float*)d_ws;
  float* h0   = ws;
  float* h1   = h0  + (size_t)BDIM * HDIM;
  float* ctx  = h1  + (size_t)BDIM * HDIM;
  float* fin  = ctx + (size_t)BDIM * CDIM;
  float* flat = fin + (size_t)BDIM * DSE;

  dim3 blk(256);
  auto gemm = [&](const float* A, const float* W, const float* bias, float* Cst,
                  int M, int N, int K, int act) {
    dim3 grid(M / BM, (N + BN - 1) / BN);
    wmma_gemm_kernel<0><<<grid, blk, 0, stream>>>(A, W, bias, Cst, M, N, K, act, se, pz);
  };

  // context_decoder: pz -> sp -> sp -> ctx
  gemm(pz,  cW1, cb1, h0,  BDIM, HDIM, CDIM, 1);
  gemm(h0,  cW2, cb2, h1,  BDIM, HDIM, HDIM, 1);
  gemm(h1,  cW3, cb3, ctx, BDIM, CDIM, HDIM, 0);

  // state_ref_encoder on virtual concat (MODE 1)
  {
    dim3 grid(BDIM / BM, (DSE + BN - 1) / BN);
    wmma_gemm_kernel<1><<<grid, blk, 0, stream>>>(nullptr, W_ref, nullptr, fin,
                                                  BDIM, DSE, TDIM * DSE, 0, se, pz);
  }

  // f_decoder: fin -> sp -> sp -> f_latent
  gemm(fin, fW1, fb1, h0,   BDIM, HDIM, DSE,  1);
  gemm(h0,  fW2, fb2, h1,   BDIM, HDIM, HDIM, 1);
  gemm(h1,  fW3, fb3, flat, BDIM, DSE,  HDIM, 0);

  // heads + log_softmax + ctx passthrough
  heads_kernel<<<dim3(BDIM), dim3(128), 0, stream>>>(
      flat, ctx, W_land, W_shot, W_move, cW_land, cW_shot, cW_move, out);

  // g-nets (dominant VALU/transcendental stage): one wave per (b,d)
  const int total = BDIM * DSE;
  gnet_kernel<<<dim3((total + 7) / 8), dim3(256), 0, stream>>>(
      flat, gW1, gb1, gW2, gb2, out, total);
}
